// HierAttNet_10050223472884
// MI455X (gfx1250) — matrix operand order repaired and checked by reading, not compiled, not run on hardware
//
#include <hip/hip_runtime.h>
#include <hip/hip_bf16.h>
#include <math.h>

// ---------------------------------------------------------------------------
// MI455X / gfx1250 implementation of the hierarchical bi-GRU + HAN network.
//  - bf16 WMMA (v_wmma_f32_16x16x32_bf16) for all GEMMs (gi precompute,
//    recurrent gh, attention projection)
//  - persistent-workgroup sequential GRU chain with LDS-resident hidden state
//  - async global->LDS double-buffered prefetch of precomputed gi rows
// ---------------------------------------------------------------------------

typedef __bf16 bf16_t;
typedef __attribute__((ext_vector_type(16))) __bf16 v16bf;
typedef __attribute__((ext_vector_type(8)))  __bf16 v8bf;
typedef __attribute__((ext_vector_type(8)))  float   v8f;

constexpr int HIDN = 256;   // per-direction hidden size (all levels)
constexpr int G3   = 768;   // 3*H gate width
constexpr int BATCH = 8;

__device__ __forceinline__ v8f wmma_bf16(v16bf a, v16bf b, v8f c) {
  // D = A(16x32) * B(32x16) + C(16x16 f32)
  return __builtin_amdgcn_wmma_f32_16x16x32_bf16(
      /*neg_a=*/false, a, /*neg_b=*/false, b,
      /*c_mod=*/(short)0, c, /*reuse_a=*/false, /*reuse_b=*/false);
}

__device__ __forceinline__ v16bf make_frag(v8bf lo, v8bf hi) {
  v16bf r;
#pragma unroll
  for (int i = 0; i < 8; ++i) { r[i] = lo[i]; r[i + 8] = hi[i]; }
  return r;
}

__device__ __forceinline__ float sigmoidf_(float x) {
  return 1.f / (1.f + expf(-x));
}

// LDS byte offset of a generic pointer to a __shared__ object
typedef __attribute__((address_space(3))) char lds_char;
__device__ __forceinline__ unsigned lds_addr_of(void* p) {
  return (unsigned)(unsigned long long)(lds_char*)p;
}

// ---------------------------------------------------------------------------
// 1) Embedding gather -> bf16 activation matrix X[(n*T+t)*B + b][E]
//    n = (p*F+f)*M+m ; matches the reference permute(1,2,3,4,0,5) layout.
// ---------------------------------------------------------------------------
__global__ void __launch_bounds__(256)
gather_embed(const int* __restrict__ tokens, const float* __restrict__ emb,
             bf16_t* __restrict__ X) {
  size_t tid = (size_t)blockIdx.x * 256 + threadIdx.x;   // 65536*256 total
  int    j   = (int)(tid & 255);
  size_t row = tid >> 8;
  int b = (int)(row & 7);
  int t = (int)((row >> 3) & 31);
  int n = (int)(row >> 8);
  int p = n >> 6, f = (n >> 3) & 7, m = n & 7;           // P=4,F=8,M=8
  int tok = tokens[((((b * 4 + p) * 8 + f) * 8 + m) * 32) + t];
  X[tid] = (bf16_t)emb[(size_t)tok * 256 + j];
}

// ---------------------------------------------------------------------------
// 2) Pack a row-major fp32 weight W[N][K] into bf16 WMMA B-fragment order:
//    out[((nt*KT+kt)*32+lane)*16 + e] = W[nt*16+(lane&15)][kt*32+(lane>>4)*8+(e&7)+(e>>3)*16]
//    (B matrix is W^T, i.e. GEMM computes A[r,k] * W[n,k].)
// ---------------------------------------------------------------------------
__global__ void __launch_bounds__(256)
pack_b(const float* __restrict__ W, bf16_t* __restrict__ out, int N, int K) {
  int tid = blockIdx.x * 256 + threadIdx.x;
  int total = N * K;
  if (tid >= total) return;
  int e    = tid & 15;
  int lane = (tid >> 4) & 31;
  int rest = tid >> 9;
  int KT = K >> 5;
  int kt = rest % KT;
  int nt = rest / KT;
  int n = nt * 16 + (lane & 15);
  int k = kt * 32 + (lane >> 4) * 8 + (e & 7) + (e >> 3) * 16;
  out[tid] = (bf16_t)W[(size_t)n * K + k];
}

// ---------------------------------------------------------------------------
// 3) Generic WMMA GEMM:  C[Rows,N] = A_bf16[Rows,K] * Bpacked + bias
//    Block = 8 waves -> 128x64 tile; wave tile = 32x32 (2x2 WMMA, A/B reuse).
// ---------------------------------------------------------------------------
__global__ void __launch_bounds__(256)
gemm_bias(const bf16_t* __restrict__ A, const bf16_t* __restrict__ Bp,
          const float* __restrict__ bias, float* __restrict__ C,
          int Rows, int K, int N) {
  const int tid  = threadIdx.x;
  const int lane = tid & 31, wave = tid >> 5;
  const int l16  = lane & 15, hk = lane >> 4;
  const int wr = wave >> 1, wc = wave & 1;
  const int r0 = blockIdx.x * 128 + wr * 32;
  const int c0 = blockIdx.y * 64 + wc * 32;
  const int nt0 = c0 >> 4;
  const int KT  = K >> 5;

  int ra0 = r0 + l16;       if (ra0 >= Rows) ra0 = 0;
  int ra1 = r0 + 16 + l16;  if (ra1 >= Rows) ra1 = 0;
  const bf16_t* arow0 = A + (size_t)ra0 * K;
  const bf16_t* arow1 = A + (size_t)ra1 * K;

  v8f acc00 = {}, acc01 = {}, acc10 = {}, acc11 = {};
  for (int kt = 0; kt < KT; ++kt) {
    const int kk = kt << 5;
    v16bf a0 = make_frag(*(const v8bf*)(arow0 + kk + hk * 8),
                         *(const v8bf*)(arow0 + kk + 16 + hk * 8));
    v16bf a1 = make_frag(*(const v8bf*)(arow1 + kk + hk * 8),
                         *(const v8bf*)(arow1 + kk + 16 + hk * 8));
    v16bf b0 = *(const v16bf*)(Bp + (((size_t)nt0 * KT + kt) * 32 + lane) * 16);
    v16bf b1 = *(const v16bf*)(Bp + (((size_t)(nt0 + 1) * KT + kt) * 32 + lane) * 16);
    acc00 = wmma_bf16(a0, b0, acc00);
    acc01 = wmma_bf16(a0, b1, acc01);
    acc10 = wmma_bf16(a1, b0, acc10);
    acc11 = wmma_bf16(a1, b1, acc11);
  }
  const float b0c = bias ? bias[c0 + l16] : 0.f;
  const float b1c = bias ? bias[c0 + 16 + l16] : 0.f;
#pragma unroll
  for (int v = 0; v < 8; ++v) {
    int rr = v + (hk ? 8 : 0);              // C layout: lanes 0-15 -> M=v, 16-31 -> M=v+8
    int row0 = r0 + rr, row1 = r0 + 16 + rr;
    int col0 = c0 + l16, col1 = c0 + 16 + l16;
    if (row0 < Rows) {
      C[(size_t)row0 * N + col0] = acc00[v] + b0c;
      C[(size_t)row0 * N + col1] = acc01[v] + b1c;
    }
    if (row1 < Rows) {
      C[(size_t)row1 * N + col0] = acc10[v] + b0c;
      C[(size_t)row1 * N + col1] = acc11[v] + b1c;
    }
  }
}

// ---------------------------------------------------------------------------
// 4) Sequential bi-GRU chain. grid.x = 2 (one block per direction).
//    Per step: gh = h * Whh^T via WMMA (h LDS-resident, Whh fragment-packed),
//    then the GRU pointwise cell.  gi rows (precomputed) stream in through
//    double-buffered async global->LDS DMA, counted with ASYNCcnt.
//    Forward writes Hout cols [0,256), backward cols [256,512) at original t.
// ---------------------------------------------------------------------------
__global__ void __launch_bounds__(256)
gru_chain(const float* __restrict__ Gi2, const bf16_t* __restrict__ WhhP2,
          const float* __restrict__ bhh2, bf16_t* __restrict__ Hout,
          int Nseq, int L, int rowsTotal) {
  __shared__ __align__(16) bf16_t h_bf[16 * HIDN];        // padded 16-row A matrix
  __shared__ __align__(16) float  h_f[BATCH * HIDN];
  __shared__ __align__(16) float  gh[BATCH * G3];
  __shared__ __align__(16) float  gi_lds[2 * BATCH * G3]; // double buffer

  const int tid  = threadIdx.x;
  const int lane = tid & 31, wave = tid >> 5;
  const int l16  = lane & 15, hk = lane >> 4;
  const int dir  = blockIdx.x;
  const float*  Gi   = Gi2 + (size_t)dir * rowsTotal * G3;
  const bf16_t* WhhP = WhhP2 + (size_t)dir * (48 * 8 * 512);
  const int colOff = dir ? HIDN : 0;

  for (int i = tid; i < 16 * HIDN; i += 256) h_bf[i] = (bf16_t)0.f;
  for (int i = tid; i < BATCH * HIDN; i += 256) h_f[i] = 0.f;

  // this thread's pointwise slice: batch row b, 8 hidden columns
  const int b  = wave;
  const int jb = lane * 8;
  float bh_r[8], bh_z[8], bh_g[8];
#pragma unroll
  for (int i = 0; i < 8; ++i) {
    bh_r[i] = bhh2[dir * G3 + jb + i];
    bh_z[i] = bhh2[dir * G3 + HIDN + jb + i];
    bh_g[i] = bhh2[dir * G3 + 2 * HIDN + jb + i];
  }
  __syncthreads();

  const unsigned giBase = lds_addr_of(gi_lds);
  const int total = Nseq * L;

  auto issue = [&](int step, int buf) {
    int n = step / L, ts = step % L;
    int t = dir ? (L - 1 - ts) : ts;
    size_t rb = ((size_t)(n * L + t) * BATCH) * G3;        // float offset
#pragma unroll
    for (int q = 0; q < 6; ++q) {                          // 1536 x 16B / 256 thr
      int c16 = tid + q * 256;
      unsigned loff = giBase + (unsigned)(buf * (BATCH * G3 * 4)) + (unsigned)c16 * 16u;
      unsigned long long ga =
          (unsigned long long)(const void*)(Gi + rb + (size_t)c16 * 4);
      asm volatile("global_load_async_to_lds_b128 %0, %1, off"
                   :: "v"(loff), "v"(ga) : "memory");
    }
  };

  issue(0, 0);

  for (int s = 0; s < total; ++s) {
    if (s + 1 < total) issue(s + 1, (s + 1) & 1);          // prefetch next step

    // ---- gh = h @ Whh^T : M=16(pad), N=768, K=256 ; wave owns 6 N-tiles ----
    v8f acc[6] = {};
    const int ntBase = wave * 6;
#pragma unroll
    for (int kt = 0; kt < 8; ++kt) {
      const bf16_t* hb = &h_bf[l16 * HIDN + kt * 32 + hk * 8];
      v16bf a = make_frag(*(const v8bf*)hb, *(const v8bf*)(hb + 16));
#pragma unroll
      for (int j = 0; j < 6; ++j) {
        const v16bf* bp = (const v16bf*)(WhhP +
            (((size_t)(ntBase + j) * 8 + kt) * 32 + lane) * 16);
        acc[j] = wmma_bf16(a, *bp, acc[j]);
      }
    }
#pragma unroll
    for (int j = 0; j < 6; ++j) {
      if (lane < 16) {                                     // only M=0..7 are real
#pragma unroll
        for (int v = 0; v < 8; ++v)
          gh[v * G3 + (ntBase + j) * 16 + lane] = acc[j][v];
      }
    }

    if (s + 1 < total) asm volatile("s_wait_asynccnt 0x6" ::: "memory");
    else               asm volatile("s_wait_asynccnt 0x0" ::: "memory");
    __syncthreads();

    // ---- pointwise GRU cell ----
    {
      int n = s / L, ts = s % L;
      int t = dir ? (L - 1 - ts) : ts;
      size_t rowBase = (size_t)(n * L + t) * BATCH;
      const float* gi = &gi_lds[(s & 1) * (BATCH * G3) + b * G3];
#pragma unroll
      for (int i = 0; i < 8; ++i) {
        int j = jb + i;
        float r  = sigmoidf_(gi[j] + gh[b * G3 + j] + bh_r[i]);
        float z  = sigmoidf_(gi[HIDN + j] + gh[b * G3 + HIDN + j] + bh_z[i]);
        float hg = gh[b * G3 + 2 * HIDN + j] + bh_g[i];
        float nn = tanhf(gi[2 * HIDN + j] + r * hg);
        float hp = h_f[b * HIDN + j];
        float hn = (1.f - z) * nn + z * hp;
        h_f[b * HIDN + j]  = hn;
        h_bf[b * HIDN + j] = (bf16_t)hn;
        Hout[(rowBase + b) * 512 + colOff + j] = (bf16_t)hn;
      }
    }
    __syncthreads();
  }
}

// ---------------------------------------------------------------------------
// 5) HAN attention pooling per sequence: scores from U (pre-tanh), softmax
//    over L, weighted sum of out -> next-level bf16 input (and fp32 repo).
// ---------------------------------------------------------------------------
__global__ void __launch_bounds__(256)
att_pool(const float* __restrict__ U, const bf16_t* __restrict__ Hout,
         const float* __restrict__ ctx, bf16_t* __restrict__ nextA,
         float* __restrict__ outF, int L) {
  __shared__ float sc[256];
  __shared__ float aw[256];
  const int tid = threadIdx.x;
  const int n = blockIdx.x;
  const int rowsSeq = L * BATCH;
  const size_t rowBase = (size_t)n * rowsSeq;

  if (tid < rowsSeq) {
    const float* u = U + (rowBase + tid) * 512;
    float s = 0.f;
    for (int g = 0; g < 512; ++g) s += tanhf(u[g]) * ctx[g];
    sc[tid] = s;
  }
  __syncthreads();
  if (tid < BATCH) {
    float m = -1e30f;
    for (int l = 0; l < L; ++l) m = fmaxf(m, sc[l * BATCH + tid]);
    float den = 0.f;
    for (int l = 0; l < L; ++l) den += expf(sc[l * BATCH + tid] - m);
    float inv = 1.f / den;
    for (int l = 0; l < L; ++l)
      aw[l * BATCH + tid] = expf(sc[l * BATCH + tid] - m) * inv;
  }
  __syncthreads();
  const int b = tid >> 5;
  const int j0 = (tid & 31) * 16;
#pragma unroll
  for (int jj = 0; jj < 16; ++jj) {
    int j = j0 + jj;
    float acc = 0.f;
    for (int l = 0; l < L; ++l)
      acc += aw[l * BATCH + b] *
             (float)Hout[(rowBase + (size_t)l * BATCH + b) * 512 + j];
    nextA[((size_t)n * BATCH + b) * 512 + j] = (bf16_t)acc;
    if (outF) outF[(size_t)b * 512 + j] = acc;
  }
}

// ---------------------------------------------------------------------------
// Host orchestration
// ---------------------------------------------------------------------------
extern "C" void kernel_launch(void* const* d_in, const int* in_sizes, int n_in,
                              void* d_out, int out_size, void* d_ws, size_t ws_size,
                              hipStream_t stream) {
  (void)in_sizes; (void)n_in; (void)out_size; (void)ws_size;
  const int*   tokens = (const int*)d_in[0];
  const float* emb    = (const float*)d_in[1];
  const float *Wih[4], *Whh[4], *bih[4], *bhh[4], *Watt[4], *batt[4], *ctx[4];
  for (int i = 0; i < 4; ++i) {
    int k = 2 + i * 7;
    Wih[i]  = (const float*)d_in[k];     Whh[i]  = (const float*)d_in[k + 1];
    bih[i]  = (const float*)d_in[k + 2]; bhh[i]  = (const float*)d_in[k + 3];
    Watt[i] = (const float*)d_in[k + 4]; batt[i] = (const float*)d_in[k + 5];
    ctx[i]  = (const float*)d_in[k + 6];
  }
  static const int DIN[4]  = {256, 512, 512, 512};
  static const int NSEQ[4] = {256, 32, 4, 1};
  static const int LSEQ[4] = {32, 8, 8, 4};
  static const int ROWS[4] = {65536, 2048, 256, 32};   // NSEQ*LSEQ*B, all %16==0

  char* ws = (char*)d_ws;
  size_t off = 0;
  auto alloc = [&](size_t bytes) -> char* {
    char* p = ws + off;
    off = (off + bytes + 255) & ~(size_t)255;
    return p;
  };
  bf16_t* X = (bf16_t*)alloc((size_t)65536 * 256 * 2);
  bf16_t *WihP[4], *WhhP[4], *WattP[4], *HoutB[4], *Anext[4];
  float  *Gi[4], *Ubuf[4];
  for (int i = 0; i < 4; ++i) {
    WihP[i]  = (bf16_t*)alloc((size_t)2 * G3 * DIN[i] * 2);
    WhhP[i]  = (bf16_t*)alloc((size_t)2 * G3 * HIDN * 2);
    WattP[i] = (bf16_t*)alloc((size_t)512 * 512 * 2);
    Gi[i]    = (float*) alloc((size_t)2 * ROWS[i] * G3 * 4);
    HoutB[i] = (bf16_t*)alloc((size_t)ROWS[i] * 512 * 2);
    Ubuf[i]  = (float*) alloc((size_t)ROWS[i] * 512 * 4);
    Anext[i] = (bf16_t*)alloc((size_t)NSEQ[i] * BATCH * 512 * 2);
  }

  // 1) embedding gather -> bf16 activations
  gather_embed<<<(65536 * 256) / 256, 256, 0, stream>>>(tokens, emb, X);

  // 2) pack all weights into WMMA B-fragment layout (bf16)
  for (int i = 0; i < 4; ++i) {
    for (int d = 0; d < 2; ++d) {
      int tW = G3 * DIN[i];
      pack_b<<<(tW + 255) / 256, 256, 0, stream>>>(
          Wih[i] + (size_t)d * tW, WihP[i] + (size_t)d * tW, G3, DIN[i]);
      int tH = G3 * HIDN;
      pack_b<<<(tH + 255) / 256, 256, 0, stream>>>(
          Whh[i] + (size_t)d * tH, WhhP[i] + (size_t)d * tH, G3, HIDN);
    }
    pack_b<<<(512 * 512 + 255) / 256, 256, 0, stream>>>(Watt[i], WattP[i], 512, 512);
  }

  // 3) per-level pipeline
  const bf16_t* Ain = X;
  for (int i = 0; i < 4; ++i) {
    // gi = A @ Wih^T + bih (both directions; fully parallel WMMA GEMM)
    dim3 gg((ROWS[i] + 127) / 128, G3 / 64);
    for (int d = 0; d < 2; ++d)
      gemm_bias<<<gg, 256, 0, stream>>>(
          Ain, WihP[i] + (size_t)d * G3 * DIN[i], bih[i] + (size_t)d * G3,
          Gi[i] + (size_t)d * ROWS[i] * G3, ROWS[i], DIN[i], G3);

    // sequential bi-GRU chain (one persistent block per direction)
    gru_chain<<<2, 256, 0, stream>>>(Gi[i], WhhP[i], bhh[i], HoutB[i],
                                     NSEQ[i], LSEQ[i], ROWS[i]);

    // U = out @ Watt^T + batt
    dim3 gu((ROWS[i] + 127) / 128, 512 / 64);
    gemm_bias<<<gu, 256, 0, stream>>>(HoutB[i], WattP[i], batt[i], Ubuf[i],
                                      ROWS[i], 512, 512);

    // attention pooling -> next level input / final repo embedding
    att_pool<<<NSEQ[i], 256, 0, stream>>>(
        Ubuf[i], HoutB[i], ctx[i], Anext[i],
        (i == 3) ? (float*)d_out : nullptr, LSEQ[i]);
    Ain = Anext[i];
  }
}
